// RNN_88046829568522
// MI455X (gfx1250) — compile-verified, matrix operand order checked by reading, not run on hardware
//
#include <hip/hip_runtime.h>
#include <hip/hip_bf16.h>

// Problem constants (match reference)
constexpr int T_ = 512, BATCH = 128, I_ = 512, H_ = 1024, O_ = 256;

typedef __attribute__((ext_vector_type(4)))  __bf16 bf16x4;
typedef __attribute__((ext_vector_type(8)))  __bf16 bf16x8;
typedef __attribute__((ext_vector_type(16))) __bf16 bf16x16;
typedef __attribute__((ext_vector_type(4)))  float  f32x4;
typedef __attribute__((ext_vector_type(8)))  float  f32x8;

// Workspace layout (bytes). Total ~132 MB.
constexpr size_t OFF_WHH = 0;                                  // H*H  bf16 = 2 MB
constexpr size_t OFF_WIH = OFF_WHH + (size_t)H_ * H_ * 2;      // H*I  bf16 = 1 MB
constexpr size_t OFF_WFC = OFF_WIH + (size_t)H_ * I_ * 2;      // O*H  bf16 = 512 KB
constexpr size_t OFF_H0  = OFF_WFC + (size_t)O_ * H_ * 2;      // B*H  bf16 = 256 KB
constexpr size_t OFF_H1  = OFF_H0  + (size_t)BATCH * H_ * 2;   // B*H  bf16 = 256 KB
constexpr size_t OFF_BAR = OFF_H1  + (size_t)BATCH * H_ * 2;   // barrier state
constexpr size_t OFF_XP  = OFF_BAR + 256;                      // T*B*H bf16 = 128 MB

__device__ __forceinline__ bf16x16 cat8(bf16x8 lo, bf16x8 hi) {
  return __builtin_shufflevector(lo, hi, 0,1,2,3,4,5,6,7,8,9,10,11,12,13,14,15);
}

// A/B fragment: two 16-byte loads (K-chunks kbase..kbase+7 and +16..+23)
__device__ __forceinline__ bf16x16 load_frag_bf16(const __bf16* __restrict__ p) {
  bf16x8 lo = *(const bf16x8*)(p);
  bf16x8 hi = *(const bf16x8*)(p + 16);
  return cat8(lo, hi);
}

// Same fragment but sourced from fp32 memory with on-the-fly bf16 convert
__device__ __forceinline__ bf16x16 load_frag_f32(const float* __restrict__ p) {
  f32x4 a = *(const f32x4*)(p);
  f32x4 b = *(const f32x4*)(p + 4);
  f32x4 c = *(const f32x4*)(p + 16);
  f32x4 d = *(const f32x4*)(p + 20);
  bf16x16 r;
  r[0]=(__bf16)a[0];  r[1]=(__bf16)a[1];  r[2]=(__bf16)a[2];  r[3]=(__bf16)a[3];
  r[4]=(__bf16)b[0];  r[5]=(__bf16)b[1];  r[6]=(__bf16)b[2];  r[7]=(__bf16)b[3];
  r[8]=(__bf16)c[0];  r[9]=(__bf16)c[1];  r[10]=(__bf16)c[2]; r[11]=(__bf16)c[3];
  r[12]=(__bf16)d[0]; r[13]=(__bf16)d[1]; r[14]=(__bf16)d[2]; r[15]=(__bf16)d[3];
  return r;
}

__device__ __forceinline__ f32x8 wmma_bf16(bf16x16 a, bf16x16 b, f32x8 c) {
  return __builtin_amdgcn_wmma_f32_16x16x32_bf16(
      /*neg_a=*/false, a, /*neg_b=*/false, b,
      /*c_mod=*/(short)0, c, /*reuse_a=*/false, /*reuse_b=*/false);
}

// ---------------- fp32 -> bf16 weight conversion ----------------
__global__ __launch_bounds__(256) void cvt_f32_bf16(
    const float* __restrict__ s, __bf16* __restrict__ d, int n) {
  int i = (blockIdx.x * 256 + threadIdx.x) * 4;
  if (i + 3 < n) {
    f32x4 v = *(const f32x4*)(s + i);
    bf16x4 o; o[0]=(__bf16)v[0]; o[1]=(__bf16)v[1]; o[2]=(__bf16)v[2]; o[3]=(__bf16)v[3];
    *(bf16x4*)(d + i) = o;
  }
}

// ---------------- x_proj = seqs @ W_ih^T + (b_ih + b_hh)  -> bf16 ----------------
// M = T*B = 65536, N = H = 1024, K = I = 512.
// Each wave computes a 16x64 strip (4 n-tiles): one fp32->bf16 A fragment feeds
// 4 WMMAs, amortizing the conversion VALU work and quartering A-side loads.
// grid: 4096 m-tiles x 2 blocks; 8 waves/block -> 16 strip slots covering N=1024.
__global__ __launch_bounds__(256) void xproj_kernel(
    const float* __restrict__ seqs, const __bf16* __restrict__ wih,
    const float* __restrict__ b_ih, const float* __restrict__ b_hh,
    __bf16* __restrict__ xp)
{
  const int lane   = threadIdx.x & 31;
  const int wave   = threadIdx.x >> 5;
  const int mtile  = blockIdx.x >> 1;
  const int ntbase = ((blockIdx.x & 1) * 8 + wave) * 4;   // first of 4 n-tiles
  const int mbase  = mtile * 16;
  const int l15    = lane & 15;
  const int koff   = (lane & 16) ? 8 : 0;

  const float*  arow = seqs + (size_t)(mbase + l15) * I_ + koff;
  const __bf16* brow = wih  + (size_t)(ntbase * 16 + l15) * I_ + koff;

  f32x8 c[4] = {{}, {}, {}, {}};
  #pragma unroll 2
  for (int k = 0; k < I_; k += 32) {
    bf16x16 a = load_frag_f32(arow + k);
    #pragma unroll
    for (int j = 0; j < 4; ++j)
      c[j] = wmma_bf16(a, load_frag_bf16(brow + (size_t)j * 16 * I_ + k), c[j]);
  }

  const int m8 = (lane >> 4) * 8;
  #pragma unroll
  for (int j = 0; j < 4; ++j) {
    const int   n    = (ntbase + j) * 16 + l15;
    const float bias = b_ih[n] + b_hh[n];
    __bf16* dst = xp + (size_t)(mbase + m8) * H_ + n;
    #pragma unroll
    for (int r = 0; r < 8; ++r)
      dst[(size_t)r * H_] = (__bf16)(c[j][r] + bias);
  }
}

// ---------------- persistent recurrent scan ----------------
// h_new = relu(x_proj[t] + h @ W_hh^T), 512 serial steps.
// 64 blocks (one n-tile each) x 8 waves (one m-tile each) = full 128x1024 output;
// keeps 256 SIMDs on the serial critical path. Block's 16-column slice of W_hh
// lives in LDS in B-fragment layout (32 KB), loaded once, reused all 512 steps.
__global__ __launch_bounds__(256) void rnn_scan(
    const __bf16* __restrict__ xp, const __bf16* __restrict__ whh,
    __bf16* __restrict__ h0, __bf16* __restrict__ h1,
    unsigned* __restrict__ bar)
{
  __shared__ __bf16 bsh[32 * 32 * 16];   // 32 K-frags x 32 lanes x 16 bf16 = 32 KB

  const int lane  = threadIdx.x & 31;
  const int wave  = threadIdx.x >> 5;
  const int mbase = wave * 16;
  const int nbase = blockIdx.x * 16;
  const int l15   = lane & 15;
  const int koff  = (lane & 16) ? 8 : 0;

  // Stage W_hh B-fragments into LDS (each wave loads 4 of the 32 K-fragments)
  {
    const __bf16* brow = whh + (size_t)(nbase + l15) * H_ + koff;
    #pragma unroll
    for (int j = 0; j < 4; ++j) {
      const int f = wave * 4 + j;
      bf16x8 lo = *(const bf16x8*)(brow + f * 32);
      bf16x8 hi = *(const bf16x8*)(brow + f * 32 + 16);
      __bf16* p = bsh + ((size_t)(f * 32 + lane)) * 16;
      *(bf16x8*)p       = lo;
      *(bf16x8*)(p + 8) = hi;
    }
  }
  __syncthreads();

  const unsigned nblocks = gridDim.x;
  const int m8 = (lane >> 4) * 8;
  const int n  = nbase + l15;

  for (int t = 0; t < T_; ++t) {
    const __bf16* hp = (t & 1) ? h1 : h0;   // read
    __bf16*       hn = (t & 1) ? h0 : h1;   // write (t=511 -> h0 holds final h)

    // Prefetch x_proj[t] values now: independent of h, latency hides behind WMMAs
    float xv[8];
    {
      const __bf16* xrow = xp + ((size_t)t * BATCH + mbase + m8) * H_ + n;
      #pragma unroll
      for (int r = 0; r < 8; ++r) xv[r] = (float)xrow[(size_t)r * H_];
    }

    const __bf16* arow = hp + (size_t)(mbase + l15) * H_ + koff;
    f32x8 c = {};
    #pragma unroll 4
    for (int f = 0; f < 32; ++f) {
      bf16x16 a = load_frag_bf16(arow + f * 32);
      const __bf16* p = bsh + ((size_t)(f * 32 + lane)) * 16;
      bf16x16 b = cat8(*(const bf16x8*)p, *(const bf16x8*)(p + 8));
      c = wmma_bf16(a, b, c);
    }

    __bf16* hw = hn + (size_t)(mbase + m8) * H_ + n;
    #pragma unroll
    for (int r = 0; r < 8; ++r) {
      float v = c[r] + xv[r];
      hw[(size_t)r * H_] = (__bf16)(v > 0.f ? v : 0.f);
    }

    // ---- device-wide barrier (monotonic count, no reset race) ----
    __threadfence();          // release h writes to device scope
    __syncthreads();
    if (threadIdx.x == 0) {
      unsigned arrived = atomicAdd(bar, 1u);
      if (arrived == nblocks * (unsigned)(t + 1) - 1u) {
        __hip_atomic_store(bar + 1, (unsigned)(t + 1),
                           __ATOMIC_RELEASE, __HIP_MEMORY_SCOPE_AGENT);
      } else {
        while (__hip_atomic_load(bar + 1, __ATOMIC_ACQUIRE,
                                 __HIP_MEMORY_SCOPE_AGENT) < (unsigned)(t + 1)) {
          __builtin_amdgcn_s_sleep(1);
        }
      }
      __threadfence();        // acquire
    }
    __syncthreads();
  }
}

// ---------------- out = h_final @ W_fc^T + b_fc  (fp32 out) ----------------
// M=128, N=256, K=1024. 8 m-tiles x 16 n-tiles = 128 tiles; 16 blocks x 8 waves.
__global__ __launch_bounds__(256) void fc_kernel(
    const __bf16* __restrict__ h, const __bf16* __restrict__ wfc,
    const float* __restrict__ b_fc, float* __restrict__ out)
{
  const int lane  = threadIdx.x & 31;
  const int wave  = threadIdx.x >> 5;
  const int mbase = wave * 16;
  const int nbase = blockIdx.x * 16;
  const int l15   = lane & 15;
  const int koff  = (lane & 16) ? 8 : 0;

  const __bf16* arow = h   + (size_t)(mbase + l15) * H_ + koff;
  const __bf16* brow = wfc + (size_t)(nbase + l15) * H_ + koff;

  f32x8 c = {};
  #pragma unroll 4
  for (int k = 0; k < H_; k += 32)
    c = wmma_bf16(load_frag_bf16(arow + k), load_frag_bf16(brow + k), c);

  const int   m8   = (lane >> 4) * 8;
  const int   n    = nbase + l15;
  const float bias = b_fc[n];
  float* dst = out + (size_t)(mbase + m8) * O_ + n;
  #pragma unroll
  for (int r = 0; r < 8; ++r)
    dst[(size_t)r * O_] = c[r] + bias;
}

extern "C" void kernel_launch(void* const* d_in, const int* in_sizes, int n_in,
                              void* d_out, int out_size, void* d_ws, size_t ws_size,
                              hipStream_t stream) {
  const float* seqs = (const float*)d_in[0];
  const float* W_ih = (const float*)d_in[1];
  const float* W_hh = (const float*)d_in[2];
  const float* b_ih = (const float*)d_in[3];
  const float* b_hh = (const float*)d_in[4];
  const float* W_fc = (const float*)d_in[5];
  const float* b_fc = (const float*)d_in[6];
  float* out = (float*)d_out;

  char* ws = (char*)d_ws;
  __bf16* whh_bf = (__bf16*)(ws + OFF_WHH);
  __bf16* wih_bf = (__bf16*)(ws + OFF_WIH);
  __bf16* wfc_bf = (__bf16*)(ws + OFF_WFC);
  __bf16* h0     = (__bf16*)(ws + OFF_H0);
  __bf16* h1     = (__bf16*)(ws + OFF_H1);
  unsigned* bar  = (unsigned*)(ws + OFF_BAR);
  __bf16* xp     = (__bf16*)(ws + OFF_XP);

  // Deterministic per-call init: zero h0 (initial hidden state) + barrier state.
  hipMemsetAsync(ws + OFF_H0, 0, (size_t)BATCH * H_ * 2, stream);
  hipMemsetAsync(ws + OFF_BAR, 0, 256, stream);

  // Weights -> bf16 (one-shot, tiny)
  cvt_f32_bf16<<<(H_ * H_ / 4 + 255) / 256, 256, 0, stream>>>(W_hh, whh_bf, H_ * H_);
  cvt_f32_bf16<<<(H_ * I_ / 4 + 255) / 256, 256, 0, stream>>>(W_ih, wih_bf, H_ * I_);
  cvt_f32_bf16<<<(O_ * H_ / 4 + 255) / 256, 256, 0, stream>>>(W_fc, wfc_bf, O_ * H_);

  // Big parallel input-projection GEMM (69 GFLOP, WMMA bf16, 16x64 strips/wave)
  xproj_kernel<<<4096 * 2, 256, 0, stream>>>(seqs, wih_bf, b_ih, b_hh, xp);

  // Persistent recurrent scan: 512 steps, grid barrier per step
  rnn_scan<<<64, 256, 0, stream>>>(xp, whh_bf, h0, h1, bar);

  // Final FC from h0 (t=511 writes h0)
  fc_kernel<<<16, 256, 0, stream>>>(h0, wfc_bf, b_fc, out);
}
